// HybridSurvivalQ_v2_9509057593683
// MI455X (gfx1250) — matrix operand order repaired and checked by reading, not compiled, not run on hardware
//
#include <hip/hip_runtime.h>

#define NQ 7
#define NL 3

typedef __attribute__((ext_vector_type(16))) _Float16 v16h;
typedef __attribute__((ext_vector_type(8)))  float    v8f;

// ------------------------------------------------------------------
// Quantum statevector kernel: one wave32 per sample.
// Amplitude index a = lane*4 + j  (j = bits[1:0], lane = bits[6:2]).
// Qubit i of the reference corresponds to bit (6-i) of the index.
// ------------------------------------------------------------------
__device__ __forceinline__ void gate_ry(float re[4], float im[4], int lane, int m, float h) {
  float s, c;
  __sincosf(h, &s, &c);
  if (m >= 4) {
    const int lm = m >> 2;
#pragma unroll
    for (int j = 0; j < 4; ++j) {
      float pr = __shfl_xor(re[j], lm, 32);
      float pi = __shfl_xor(im[j], lm, 32);
      float sg = (lane & lm) ? s : -s;
      re[j] = c * re[j] + sg * pr;
      im[j] = c * im[j] + sg * pi;
    }
  } else {
#pragma unroll
    for (int j = 0; j < 4; ++j) {
      if ((j & m) == 0) {
        int j1 = j | m;
        float r0 = re[j], i0 = im[j], r1 = re[j1], i1 = im[j1];
        re[j]  = c * r0 - s * r1;  im[j]  = c * i0 - s * i1;
        re[j1] = s * r0 + c * r1;  im[j1] = s * i0 + c * i1;
      }
    }
  }
}

__device__ __forceinline__ void gate_rz(float re[4], float im[4], int lane, int m, float h) {
  float sh, ch;
  __sincosf(h, &sh, &ch);
#pragma unroll
  for (int j = 0; j < 4; ++j) {
    bool bit = (m >= 4) ? ((lane & (m >> 2)) != 0) : ((j & m) != 0);
    float sw = bit ? sh : -sh;
    float r = re[j], i = im[j];
    re[j] = r * ch - i * sw;
    im[j] = r * sw + i * ch;
  }
}

__device__ __forceinline__ void gate_cnot(float re[4], float im[4], int lane, int mc, int mt) {
  if (mt >= 4) {
    const int lm = mt >> 2;
#pragma unroll
    for (int j = 0; j < 4; ++j) {
      float pr = __shfl_xor(re[j], lm, 32);
      float pi = __shfl_xor(im[j], lm, 32);
      bool cb = (mc >= 4) ? ((lane & (mc >> 2)) != 0) : ((j & mc) != 0);
      re[j] = cb ? pr : re[j];
      im[j] = cb ? pi : im[j];
    }
  } else {
#pragma unroll
    for (int j = 0; j < 4; ++j) {
      if ((j & mt) == 0) {
        int j1 = j | mt;
        bool cb = (mc >= 4) ? ((lane & (mc >> 2)) != 0) : ((j & mc) != 0);
        float r0 = re[j], r1 = re[j1], i0 = im[j], i1 = im[j1];
        re[j]  = cb ? r1 : r0;  im[j]  = cb ? i1 : i0;
        re[j1] = cb ? r0 : r1;  im[j1] = cb ? i0 : i1;
      }
    }
  }
}

__global__ __launch_bounds__(256) void quantum_kernel(
    const float* __restrict__ x_q, const float* __restrict__ q_ry,
    const float* __restrict__ q_rz, float* __restrict__ q_out, int B) {
  const int lane = threadIdx.x & 31;
  const int wv   = threadIdx.x >> 5;
  const int s    = blockIdx.x * 8 + wv;
  if (s >= B) return;  // uniform per wave

  float re[4], im[4];
#pragma unroll
  for (int j = 0; j < 4; ++j) { re[j] = 0.f; im[j] = 0.f; }
  if (lane == 0) re[0] = 1.0f;  // |0...0>

  const float* xq = x_q + (size_t)s * NQ;

  for (int layer = 0; layer < NL; ++layer) {
    // data re-uploading RY(x_q[s,i]) on qubit i
#pragma unroll
    for (int i = 0; i < NQ; ++i)
      gate_ry(re, im, lane, 1 << (6 - i), xq[i] * 0.5f);
    // trained RY, RZ
#pragma unroll
    for (int i = 0; i < NQ; ++i) {
      gate_ry(re, im, lane, 1 << (6 - i), q_ry[layer * NQ + i] * 0.5f);
      gate_rz(re, im, lane, 1 << (6 - i), q_rz[layer * NQ + i] * 0.5f);
    }
    const int shift = layer + 1;
#pragma unroll
    for (int i = 0; i < NQ; ++i)
      gate_cnot(re, im, lane, 1 << (6 - i), 1 << (6 - ((i + shift) % NQ)));
  }

  // Z expectations per qubit
  float zp[NQ];
#pragma unroll
  for (int i = 0; i < NQ; ++i) {
    const int m = 1 << (6 - i);
    float acc = 0.f;
#pragma unroll
    for (int j = 0; j < 4; ++j) {
      int a = lane * 4 + j;
      float p = re[j] * re[j] + im[j] * im[j];
      acc += (a & m) ? -p : p;
    }
    zp[i] = acc;
  }
#pragma unroll
  for (int off = 16; off >= 1; off >>= 1) {
#pragma unroll
    for (int i = 0; i < NQ; ++i) zp[i] += __shfl_xor(zp[i], off, 32);
  }
  if (lane == 0) {
#pragma unroll
    for (int i = 0; i < NQ; ++i) q_out[(size_t)s * NQ + i] = zp[i];
  }
}

// ------------------------------------------------------------------
// Fused classical MLP + head, WMMA f16->f32. 8 waves/block, each wave
// owns 16 rows. Weights live transposed ([n][k], f16) in LDS.
// Fragment layouts (wave32, 16x16x32 f16 WMMA):
//   A (16x32):  lane l: m = l&15;  k = 2*(v&3) + 16*(v>>2) + 8*(l>>4) + p
//   B (32x16):  lane l: n = l&15;  k = 2*(v&3) + 16*(v>>2) + 8*(l>>4) + p
//   C/D (16x16): lane l, reg r: m = r + 8*(l>>4), n = l&15
// ------------------------------------------------------------------
__device__ __forceinline__ int kA(int v, int hi) {
  return ((v & 3) << 1) + ((v >> 2) << 4) + (hi << 3);
}
__device__ __forceinline__ int kB(int v, int hi) {
  return ((v & 3) << 1) + ((v >> 2) << 4) + (hi << 3);
}
__device__ __forceinline__ v8f wmma_f16(const v16h a, const v16h b, v8f c) {
  return __builtin_amdgcn_wmma_f32_16x16x32_f16(false, a, false, b, (short)0, c,
                                                false, false);
}

__global__ __launch_bounds__(256) void mlp_kernel(
    const float* __restrict__ x_c,
    const float* __restrict__ We1, const float* __restrict__ be1,
    const float* __restrict__ We2, const float* __restrict__ be2,
    const float* __restrict__ Wf1, const float* __restrict__ bf1,
    const float* __restrict__ Wf2, const float* __restrict__ bf2,
    const float* __restrict__ Wf3, const float* __restrict__ bf3,
    const float* __restrict__ q_out, float* __restrict__ out, int B) {
  __shared__ _Float16 sWe1[64 * 128];   // [n][k] transposed
  __shared__ _Float16 sWe2[32 * 64];    // [n][k]
  __shared__ _Float16 sWf1[64 * 64];    // [n][k], k>=39 zero-padded
  __shared__ _Float16 sWf2[32 * 64];    // [n][k]
  __shared__ _Float16 sStage[8][16 * 64];  // per-wave activation staging

  const int tid = threadIdx.x;
  for (int idx = tid; idx < 64 * 128; idx += 256) {
    int n = idx >> 7, k = idx & 127;
    sWe1[idx] = (_Float16)We1[k * 64 + n];
  }
  for (int idx = tid; idx < 32 * 64; idx += 256) {
    int n = idx >> 6, k = idx & 63;
    sWe2[idx] = (_Float16)We2[k * 32 + n];
  }
  for (int idx = tid; idx < 64 * 64; idx += 256) {
    int n = idx >> 6, k = idx & 63;
    sWf1[idx] = (k < 39) ? (_Float16)Wf1[k * 64 + n] : (_Float16)0.f;
  }
  for (int idx = tid; idx < 32 * 64; idx += 256) {
    int n = idx >> 6, k = idx & 63;
    sWf2[idx] = (_Float16)Wf2[k * 32 + n];
  }
  __syncthreads();

  const int lane = tid & 31;
  const int wv   = tid >> 5;
  const int hi   = lane >> 4;
  const int nl   = lane & 15;
  const int row0 = blockIdx.x * 128 + wv * 16;
  _Float16* stage = sStage[wv];

  // ---------------- layer e1: x_c(16x128) @ We1(128x64) ----------------
  v8f acc1[4];
#pragma unroll
  for (int t = 0; t < 4; ++t)
#pragma unroll
    for (int r = 0; r < 8; ++r) acc1[t][r] = 0.f;

  const float* arow = x_c + (size_t)(row0 + nl) * 128;  // A row m = nl
#pragma unroll
  for (int kc = 0; kc < 4; ++kc) {
    v16h a;
#pragma unroll
    for (int v = 0; v < 8; ++v) {
      float2 f = *(const float2*)(arow + kc * 32 + kA(v, hi));
      a[2 * v]     = (_Float16)f.x;
      a[2 * v + 1] = (_Float16)f.y;
    }
#pragma unroll
    for (int t = 0; t < 4; ++t) {
      const int n_g = t * 16 + nl;
      v16h b;
#pragma unroll
      for (int v = 0; v < 8; ++v) {
        int k = kc * 32 + kB(v, hi);
        b[2 * v]     = sWe1[n_g * 128 + k];
        b[2 * v + 1] = sWe1[n_g * 128 + k + 1];
      }
      acc1[t] = wmma_f16(a, b, acc1[t]);
    }
  }
  // relu + bias -> stage h (16x64 f16)
#pragma unroll
  for (int t = 0; t < 4; ++t) {
    const int n_g = t * 16 + nl;
    const float bb = be1[n_g];
#pragma unroll
    for (int r = 0; r < 8; ++r) {
      int m = r + 8 * hi;
      float hv = acc1[t][r] + bb;
      stage[m * 64 + n_g] = (_Float16)(hv > 0.f ? hv : 0.f);
    }
  }

  // ---------------- layer e2: h(16x64) @ We2(64x32) ----------------
  v8f acc2[2];
#pragma unroll
  for (int t = 0; t < 2; ++t)
#pragma unroll
    for (int r = 0; r < 8; ++r) acc2[t][r] = 0.f;
#pragma unroll
  for (int kc = 0; kc < 2; ++kc) {
    v16h a;
#pragma unroll
    for (int v = 0; v < 8; ++v) {
      int k = kc * 32 + kA(v, hi);
      a[2 * v]     = stage[nl * 64 + k];
      a[2 * v + 1] = stage[nl * 64 + k + 1];
    }
#pragma unroll
    for (int t = 0; t < 2; ++t) {
      const int n_g = t * 16 + nl;
      v16h b;
#pragma unroll
      for (int v = 0; v < 8; ++v) {
        int k = kc * 32 + kB(v, hi);
        b[2 * v]     = sWe2[n_g * 64 + k];
        b[2 * v + 1] = sWe2[n_g * 64 + k + 1];
      }
      acc2[t] = wmma_f16(a, b, acc2[t]);
    }
  }

  // ---------------- build z = [q_out(7) | c_out(32) | 0-pad] in stage ----------------
#pragma unroll
  for (int t = 0; t < 2; ++t) {
    const int n_g = t * 16 + nl;
    const float bb = be2[n_g];
#pragma unroll
    for (int r = 0; r < 8; ++r) {
      int m = r + 8 * hi;
      float cv = acc2[t][r] + bb;
      stage[m * 64 + 7 + n_g] = (_Float16)(cv > 0.f ? cv : 0.f);
    }
  }
  if (hi == 0) {  // lane handles row m = nl
    const float* qrow = q_out + (size_t)(row0 + nl) * NQ;
#pragma unroll
    for (int i = 0; i < NQ; ++i) stage[nl * 64 + i] = (_Float16)qrow[i];
#pragma unroll
    for (int c = 39; c < 64; ++c) stage[nl * 64 + c] = (_Float16)0.f;
  }

  // ---------------- layer f1: z(16x64) @ Wf1pad(64x64) ----------------
  v8f acc3[4];
#pragma unroll
  for (int t = 0; t < 4; ++t)
#pragma unroll
    for (int r = 0; r < 8; ++r) acc3[t][r] = 0.f;
#pragma unroll
  for (int kc = 0; kc < 2; ++kc) {
    v16h a;
#pragma unroll
    for (int v = 0; v < 8; ++v) {
      int k = kc * 32 + kA(v, hi);
      a[2 * v]     = stage[nl * 64 + k];
      a[2 * v + 1] = stage[nl * 64 + k + 1];
    }
#pragma unroll
    for (int t = 0; t < 4; ++t) {
      const int n_g = t * 16 + nl;
      v16h b;
#pragma unroll
      for (int v = 0; v < 8; ++v) {
        int k = kc * 32 + kB(v, hi);
        b[2 * v]     = sWf1[n_g * 64 + k];
        b[2 * v + 1] = sWf1[n_g * 64 + k + 1];
      }
      acc3[t] = wmma_f16(a, b, acc3[t]);
    }
  }
  // relu + bf1 -> stage z2 (16x64)
#pragma unroll
  for (int t = 0; t < 4; ++t) {
    const int n_g = t * 16 + nl;
    const float bb = bf1[n_g];
#pragma unroll
    for (int r = 0; r < 8; ++r) {
      int m = r + 8 * hi;
      float zv = acc3[t][r] + bb;
      stage[m * 64 + n_g] = (_Float16)(zv > 0.f ? zv : 0.f);
    }
  }

  // ---------------- layer f2: z2(16x64) @ Wf2(64x32) ----------------
  v8f acc4[2];
#pragma unroll
  for (int t = 0; t < 2; ++t)
#pragma unroll
    for (int r = 0; r < 8; ++r) acc4[t][r] = 0.f;
#pragma unroll
  for (int kc = 0; kc < 2; ++kc) {
    v16h a;
#pragma unroll
    for (int v = 0; v < 8; ++v) {
      int k = kc * 32 + kA(v, hi);
      a[2 * v]     = stage[nl * 64 + k];
      a[2 * v + 1] = stage[nl * 64 + k + 1];
    }
#pragma unroll
    for (int t = 0; t < 2; ++t) {
      const int n_g = t * 16 + nl;
      v16h b;
#pragma unroll
      for (int v = 0; v < 8; ++v) {
        int k = kc * 32 + kB(v, hi);
        b[2 * v]     = sWf2[n_g * 64 + k];
        b[2 * v + 1] = sWf2[n_g * 64 + k + 1];
      }
      acc4[t] = wmma_f16(a, b, acc4[t]);
    }
  }

  // ---------------- layer f3: relu(z3) @ Wf3(32x1) + bf3 ----------------
  float part[8];
#pragma unroll
  for (int r = 0; r < 8; ++r) part[r] = 0.f;
#pragma unroll
  for (int t = 0; t < 2; ++t) {
    const int n_g = t * 16 + nl;
    const float bb = bf2[n_g];
    const float w3 = Wf3[n_g];
#pragma unroll
    for (int r = 0; r < 8; ++r) {
      float zv = acc4[t][r] + bb;
      zv = zv > 0.f ? zv : 0.f;
      part[r] += zv * w3;
    }
  }
  // reduce over the 16 lanes of each half (n dimension)
#pragma unroll
  for (int off = 1; off < 16; off <<= 1) {
#pragma unroll
    for (int r = 0; r < 8; ++r) part[r] += __shfl_xor(part[r], off, 32);
  }
  if (nl == 0) {
    const float b3 = bf3[0];
#pragma unroll
    for (int r = 0; r < 8; ++r) out[row0 + r + 8 * hi] = part[r] + b3;
  }
}

// ------------------------------------------------------------------
extern "C" void kernel_launch(void* const* d_in, const int* in_sizes, int n_in,
                              void* d_out, int out_size, void* d_ws, size_t ws_size,
                              hipStream_t stream) {
  const float* x_q  = (const float*)d_in[0];
  const float* x_c  = (const float*)d_in[1];
  const float* q_ry = (const float*)d_in[2];
  const float* q_rz = (const float*)d_in[3];
  const float* We1  = (const float*)d_in[4];
  const float* be1  = (const float*)d_in[5];
  const float* We2  = (const float*)d_in[6];
  const float* be2  = (const float*)d_in[7];
  const float* Wf1  = (const float*)d_in[8];
  const float* bf1  = (const float*)d_in[9];
  const float* Wf2  = (const float*)d_in[10];
  const float* bf2  = (const float*)d_in[11];
  const float* Wf3  = (const float*)d_in[12];
  const float* bf3  = (const float*)d_in[13];
  float* out = (float*)d_out;

  const int B = in_sizes[0] / NQ;  // 65536
  float* q_out = (float*)d_ws;     // B*7 floats scratch

  quantum_kernel<<<(B + 7) / 8, 256, 0, stream>>>(x_q, q_ry, q_rz, q_out, B);
  mlp_kernel<<<B / 128, 256, 0, stream>>>(x_c, We1, be1, We2, be2, Wf1, bf1,
                                          Wf2, bf2, Wf3, bf3, q_out, out, B);
}
